// ARNN_50740743635547
// MI455X (gfx1250) — matrix operand order repaired and checked by reading, not compiled, not run on hardware
//
#include <hip/hip_runtime.h>
#include <hip/hip_bf16.h>

// ---------------------------------------------------------------------------
// ACT-RNN persistent-cell kernel for MI455X (gfx1250, wave32, WMMA).
//
// One workgroup (256 threads = 8 wave32) owns a 16-row batch tile and runs
// the full 256-timestep x (<=21)-ponder-step recursion with all weights
// resident in LDS as f16 (row-major [N][K] == native WMMA B-fragment image).
// Recurrent state s lives in LDS as [16][256] f16 (native A-fragment image).
// x_{t+1} is prefetched (global_prefetch_b8) while timestep t ponders.
// ---------------------------------------------------------------------------

typedef _Float16 h16;
typedef __attribute__((ext_vector_type(16))) _Float16 v16h;
typedef __attribute__((ext_vector_type(8)))  _Float16 v8h;
typedef __attribute__((ext_vector_type(8)))  float    v8f;

#define Tn 256
#define Bn 512
#define In 128
#define Hn 256
#define On 128
#define BT 16          // batch tile (WMMA M)
#define TPB 256        // 8 waves
#define MAXP 20        // ponder steps 0..20 inclusive

__device__ __forceinline__ v16h cat16(v8h lo, v8h hi) {
  return __builtin_shufflevector(lo, hi, 0,1,2,3,4,5,6,7,8,9,10,11,12,13,14,15);
}

// A-fragment (16x32 f16, MxK) from row-major [M][rs] f16 LDS tile.
// lanes 0-15: row=lane, K = k0+0..7 (v0-3) and k0+16..23 (v4-7)
// lanes16-31: row=lane-16, K = k0+8..15 and k0+24..31
__device__ __forceinline__ v16h load_afrag(const h16* base, int rs, int lane, int k0) {
  int row  = lane & 15;
  int koff = (lane >> 4) << 3;
  const h16* p = base + row * rs + k0 + koff;
  v8h lo = *(const v8h*)p;
  v8h hi = *(const v8h*)(p + 16);
  return cat16(lo, hi);
}

// B-fragment (32x16 f16, KxN) from row-major [N][rs] f16 LDS weight tile.
// lane n (<16): col=colbase+n, K=k0..k0+7,k0+16..23 ; lanes 16-31: K=k0+8..15,k0+24..31
__device__ __forceinline__ v16h load_bfrag(const h16* base, int rs, int lane,
                                           int colbase, int k0) {
  int col = colbase + (lane & 15);
  int kb  = k0 + ((lane >> 4) << 3);
  const h16* p = base + col * rs + kb;
  v8h lo = *(const v8h*)p;
  v8h hi = *(const v8h*)(p + 16);
  return cat16(lo, hi);
}

__device__ __forceinline__ v8f wmma16(v16h a, v16h b, v8f c) {
  return __builtin_amdgcn_wmma_f32_16x16x32_f16(false, a, false, b,
                                                (short)0, c, false, false);
}

__global__ __launch_bounds__(TPB) void act_rnn_persistent(
    const float* __restrict__ xg,     const float* __restrict__ s0g,
    const float* __restrict__ Wih_g,  const float* __restrict__ bih_g,
    const float* __restrict__ Whh_g,  const float* __restrict__ bhh_g,
    const float* __restrict__ Whalt_g,const float* __restrict__ bhalt_g,
    const float* __restrict__ Wout_g, const float* __restrict__ bout_g,
    float* __restrict__ outg, float* __restrict__ pondg) {
  extern __shared__ __align__(16) char smem[];
  h16* WhhL  = (h16*)smem;          // [256][256] = 65536 halfs
  h16* WoutL = WhhL  + Hn * Hn;     // [128][256] = 32768
  h16* Wih1L = WoutL + On * Hn;     // [256][128] = 32768
  h16* xl    = Wih1L + Hn * In;     // [16][128]  = 2048
  h16* sl    = xl    + BT * In;     // [16][256]  = 4096
  float* Wih0L  = (float*)(sl + BT * Hn); // 256
  float* biasL  = Wih0L  + Hn;            // 256 (b_ih + b_hh)
  float* WhaltL = biasL  + Hn;            // 256
  float* BoutL  = WhaltL + Hn;            // 128
  float* partL  = BoutL  + On;            // 256 halt partials
  float* haltedL= partL  + 256;           // 16
  float* haltaL = haltedL + BT;           // 16
  float* cumpL  = haltaL  + BT;           // 16
  float* pondL  = cumpL   + BT;           // 16
  float* wlL    = pondL   + BT;           // 16 (per-row weight this step)
  float* nhlL   = wlL     + BT;           // 16 (not-halted mask)
  float* bhaltL = nhlL    + BT;           // 1
  int*   allhL  = (int*)(bhaltL + 1);     // 1

  const int tid  = threadIdx.x;
  const int lane = tid & 31;
  const int wave = tid >> 5;
  const int lrow = lane & 15;
  const int hi8  = (lane >> 4) << 3;      // row offset of C-fragment half
  const int b0   = blockIdx.x * BT;
  const int col0 = wave * 32 + lrow;      // hidden columns owned (tile 0)
  const int col1 = col0 + 16;             // tile 1
  const int colo = wave * 16 + lrow;      // output column owned

  // ---- one-time weight staging (f32 -> f16 LDS) ----
  for (int r = 0; r < (Hn * Hn) / TPB; ++r) {
    int idx = tid + r * TPB; WhhL[idx] = (h16)Whh_g[idx];
  }
  for (int r = 0; r < (On * Hn) / TPB; ++r) {
    int idx = tid + r * TPB; WoutL[idx] = (h16)Wout_g[idx];
  }
  for (int r = 0; r < (Hn * In) / TPB; ++r) {
    int idx = tid + r * TPB; int h = idx >> 7, i = idx & 127;
    Wih1L[idx] = (h16)Wih_g[h * (In + 1) + 1 + i];
  }
  Wih0L[tid]  = Wih_g[tid * (In + 1)];
  biasL[tid]  = bih_g[tid] + bhh_g[tid];
  WhaltL[tid] = Whalt_g[tid];
  if (tid < On) BoutL[tid] = bout_g[tid];
  if (tid == 0) bhaltL[0] = bhalt_g[0];
  for (int r = 0; r < (BT * Hn) / TPB; ++r) {
    int idx = tid + r * TPB; int m = idx >> 8, h = idx & 255;
    sl[idx] = (h16)s0g[(size_t)(b0 + m) * Hn + h];
  }
  __syncthreads();

  for (int t = 0; t < Tn; ++t) {
    // ---- stage x_t tile (f32 global -> f16 LDS, A-image) ----
#pragma unroll
    for (int r = 0; r < (BT * In) / TPB; ++r) {
      int idx = tid + r * TPB; int m = idx >> 7, i = idx & 127;
      xl[idx] = (h16)xg[((size_t)t * Bn + b0 + m) * In + i];
    }
    // ---- prefetch x_{t+1} tile into cache (one lane per 128B line) ----
    if (t + 1 < Tn) {
#pragma unroll
      for (int r = 0; r < (BT * In) / TPB; ++r) {
        int idx = tid + r * TPB;
        if ((idx & 31) == 0) {
          int m = idx >> 7, i = idx & 127;
          __builtin_prefetch(&xg[((size_t)(t + 1) * Bn + b0 + m) * In + i], 0, 3);
        }
      }
    }
    __syncthreads();

    // ---- per-timestep precompute: xb = x_t @ W_ih[:,1:]^T + (b_ih+b_hh) ----
    v8f xb0, xb1;
    {
      float bi0 = biasL[col0], bi1 = biasL[col1];
#pragma unroll
      for (int j = 0; j < 8; ++j) { xb0[j] = bi0; xb1[j] = bi1; }
#pragma unroll
      for (int k = 0; k < In / 32; ++k) {
        v16h xa = load_afrag(xl, In, lane, k * 32);
        xb0 = wmma16(xa, load_bfrag(Wih1L, In, lane, wave * 32,      k * 32), xb0);
        xb1 = wmma16(xa, load_bfrag(Wih1L, In, lane, wave * 32 + 16, k * 32), xb1);
      }
    }
    const float w0a = Wih0L[col0], w0b = Wih0L[col1];

    v8f sacc0 = {0,0,0,0,0,0,0,0}, sacc1 = {0,0,0,0,0,0,0,0};
    v8f oacc  = {0,0,0,0,0,0,0,0};

    // ---- ponder loop ----
    for (int n = 0; n <= MAXP; ++n) {
      // phase 1: s_new = tanh(xb + s_prev @ W_hh^T)
      v16h af[8];
#pragma unroll
      for (int k = 0; k < Hn / 32; ++k) af[k] = load_afrag(sl, Hn, lane, k * 32);
      __syncthreads();  // all reads of s_prev complete before overwrite

      v8f c0 = xb0, c1 = xb1;
      if (n == 0) {
#pragma unroll
        for (int j = 0; j < 8; ++j) { c0[j] += w0a; c1[j] += w0b; }  // flag column
      }
#pragma unroll
      for (int k = 0; k < Hn / 32; ++k) {
        c0 = wmma16(af[k], load_bfrag(WhhL, Hn, lane, wave * 32,      k * 32), c0);
        c1 = wmma16(af[k], load_bfrag(WhhL, Hn, lane, wave * 32 + 16, k * 32), c1);
      }
      float sv0[8], sv1[8];
#pragma unroll
      for (int j = 0; j < 8; ++j) { sv0[j] = tanhf(c0[j]); sv1[j] = tanhf(c1[j]); }
#pragma unroll
      for (int j = 0; j < 8; ++j) {
        sl[(j + hi8) * Hn + col0] = (h16)sv0[j];
        sl[(j + hi8) * Hn + col1] = (h16)sv1[j];
      }
      __syncthreads();  // unmasked s_new visible

      // phase 2: halt partial dot products (256 threads: 16 b x 16 chunks)
      {
        int b = tid >> 4, c = tid & 15;
        float acc = 0.f;
#pragma unroll
        for (int u = 0; u < 16; ++u) {
          int h = c * 16 + u;
          acc += (float)sl[b * Hn + h] * WhaltL[h];
        }
        partL[tid] = acc;
      }
      __syncthreads();

      // ACT scalar recurrence: one lane per batch row
      float myhv = 1.f;
      if (tid < BT) {
        float z = bhaltL[0];
#pragma unroll
        for (int c = 0; c < 16; ++c) z += partL[tid * 16 + c];
        float praw = 1.f / (1.f + __expf(-z));
        float w, nhv;
        if (n == 0) {
          float tmp = (praw >= 0.99f) ? 1.f : 0.f;
          pondL[tid]  = 2.f * tmp;
          float p     = fmaxf(praw, tmp);
          haltedL[tid] = tmp;
          haltaL[tid]  = praw;
          cumpL[tid]   = p;
          w = p; nhv = 1.f; myhv = tmp;
        } else {
          float hv  = haltedL[tid];
          nhv = 1.f - hv;
          float tmp = ((nhv * (haltaL[tid] + praw)) >= 0.99f) ? 1.f : 0.f;
          float r   = tmp * (1.f - cumpL[tid]);
          pondL[tid] += (r + (float)(n + 1)) * tmp;
          float pn = praw * (1.f - tmp) + r;
          haltaL[tid] += pn;
          pn *= nhv;
          hv += tmp;
          haltedL[tid] = hv;
          cumpL[tid]  += pn;
          w = pn; myhv = hv;
        }
        wlL[tid] = w; nhlL[tid] = nhv;
      }
      if (wave == 0) {  // wave-uniform all-halted detection
        unsigned long long bal = __ballot(myhv > 0.5f);
        if (lane == 0) allhL[0] = (((unsigned)bal) == 0xffffffffu) ? 1 : 0;
      }
      __syncthreads();
      const int stop = allhL[0];

      // phase 3: out = s_new @ W_out^T + b_out ; o_acc += w[row] * out
#pragma unroll
      for (int k = 0; k < Hn / 32; ++k) af[k] = load_afrag(sl, Hn, lane, k * 32);
      v8f oc;
      {
        float bo = BoutL[colo];
#pragma unroll
        for (int j = 0; j < 8; ++j) oc[j] = bo;
      }
#pragma unroll
      for (int k = 0; k < Hn / 32; ++k)
        oc = wmma16(af[k], load_bfrag(WoutL, Hn, lane, wave * 16, k * 32), oc);

      float wrow[8];
#pragma unroll
      for (int j = 0; j < 8; ++j) wrow[j] = wlL[j + hi8];
#pragma unroll
      for (int j = 0; j < 8; ++j) oacc[j] += wrow[j] * oc[j];

      // phase 4: mask state, accumulate s_acc, carry masked state forward
#pragma unroll
      for (int j = 0; j < 8; ++j) {
        float nh = nhlL[j + hi8];
        sv0[j] *= nh; sv1[j] *= nh;                    // s_n = s_n * (1-halted)
        sacc0[j] += wrow[j] * sv0[j];
        sacc1[j] += wrow[j] * sv1[j];
      }
      __syncthreads();  // phase-3 A reads done before overwrite
#pragma unroll
      for (int j = 0; j < 8; ++j) {
        sl[(j + hi8) * Hn + col0] = (h16)sv0[j];
        sl[(j + hi8) * Hn + col1] = (h16)sv1[j];
      }
      __syncthreads();
      if (stop) break;  // all rows halted: remaining steps are exact no-ops
    }

    // ---- emit outputs, install s_carry = s_acc ----
#pragma unroll
    for (int j = 0; j < 8; ++j) {
      int m = j + hi8;
      outg[((size_t)t * Bn + b0 + m) * On + colo] = oacc[j];
    }
    if (tid < BT) pondg[(size_t)t * Bn + b0 + tid] = pondL[tid];
#pragma unroll
    for (int j = 0; j < 8; ++j) {
      sl[(j + hi8) * Hn + col0] = (h16)sacc0[j];
      sl[(j + hi8) * Hn + col1] = (h16)sacc1[j];
    }
    // next iteration's x-stage __syncthreads orders these stores before reads
  }
}

static constexpr size_t SMEM_BYTES =
    (size_t)(Hn * Hn + On * Hn + Hn * In + BT * In + BT * Hn) * sizeof(h16) +
    (size_t)(Hn * 3 + On + 256 + 6 * BT + 1) * sizeof(float) + 64;

extern "C" void kernel_launch(void* const* d_in, const int* in_sizes, int n_in,
                              void* d_out, int out_size, void* d_ws, size_t ws_size,
                              hipStream_t stream) {
  (void)in_sizes; (void)n_in; (void)out_size; (void)d_ws; (void)ws_size;
  const float* xg     = (const float*)d_in[0];
  const float* s0g    = (const float*)d_in[1];
  const float* Wih    = (const float*)d_in[2];
  const float* bih    = (const float*)d_in[3];
  const float* Whh    = (const float*)d_in[4];
  const float* bhh    = (const float*)d_in[5];
  const float* Whalt  = (const float*)d_in[6];
  const float* bhalt  = (const float*)d_in[7];
  const float* Wout   = (const float*)d_in[8];
  const float* bout   = (const float*)d_in[9];
  float* outg  = (float*)d_out;
  float* pondg = outg + (size_t)Tn * Bn * On;

  hipFuncSetAttribute((const void*)act_rnn_persistent,
                      hipFuncAttributeMaxDynamicSharedMemorySize,
                      (int)SMEM_BYTES);
  act_rnn_persistent<<<Bn / BT, TPB, SMEM_BYTES, stream>>>(
      xg, s0g, Wih, bih, Whh, bhh, Whalt, bhalt, Wout, bout, outg, pondg);
}
